// QLoRALinear_36704790511804
// MI455X (gfx1250) — matrix operand order repaired and checked by reading, not compiled, not run on hardware
//
#include <hip/hip_runtime.h>
#include <hip/hip_bf16.h>

typedef __attribute__((ext_vector_type(16))) _Float16     v16h;
typedef __attribute__((ext_vector_type(8)))  float        v8f;
typedef __attribute__((ext_vector_type(4)))  unsigned int v4u;
typedef __attribute__((ext_vector_type(8)))  int          v8i;
typedef __attribute__((ext_vector_type(4)))  int          v4i;

#define M_DIM   128
#define K_DIM   4096
#define N_DIM   11008
#define RLORA   16
#define SCALING 2.0f

#define NB      64          // N columns per block
#define KC      64          // K chunk (half a quant group)
#define NCHUNK  (K_DIM / KC)
#define XSTR    72          // padded LDS stride (halves)
#define WSTR    72
#define RAWELEM (KC * NB)   // raw int32 codes per chunk

// ---- WMMA fragment gathers from LDS (layouts per CDNA5 ISA 7.12.2) ----
__device__ __forceinline__ v16h frag_a(const _Float16* rowp, int kk, int half) {
    union { v16h v; uint4 q[2]; } u;
    u.q[0] = *(const uint4*)(rowp + kk + 8 * half);
    u.q[1] = *(const uint4*)(rowp + kk + 16 + 8 * half);
    return u.v;
}
__device__ __forceinline__ v16h frag_b(const _Float16* colp, int kk, int khalf) {
    union { v16h v; uint4 q[2]; } u;
    const _Float16* p = colp + kk + 8 * khalf;
    u.q[0] = *(const uint4*)(p);
    u.q[1] = *(const uint4*)(p + 16);
    return u.v;
}

// ---- TDM: DMA one 64(k) x 64(n) int32 tile of qw into LDS (2-D descriptor) ----
__device__ __forceinline__ void tdm_load_w(const int* qw, int k0, int n0, unsigned lds_off) {
    unsigned long long ga = (unsigned long long)(const void*)qw +
                            ((unsigned long long)k0 * N_DIM + (unsigned)n0) * 4ull;
    v4u g0;
    g0.x = 1u;                                            // count=1, user descriptor
    g0.y = lds_off;                                       // lds_addr (bytes)
    g0.z = (unsigned)(ga & 0xFFFFFFFFull);                // global_addr[31:0]
    g0.w = (unsigned)((ga >> 32) & 0x1FFFFFFull) | 0x80000000u;  // addr[56:32] | type=2
    v8i g1;
    g1[0] = 0x20000;                                      // data_size=2 (4B)
    g1[1] = (int)((N_DIM & 0xFFFF) << 16);                // tensor_dim0[15:0]
    g1[2] = (int)(((N_DIM >> 16) & 0xFFFF) | ((K_DIM & 0xFFFF) << 16)); // dim0 hi | dim1 lo
    g1[3] = (int)(((K_DIM >> 16) & 0xFFFF) | (NB << 16)); // dim1 hi | tile_dim0=64
    g1[4] = KC;                                           // tile_dim1=64, tile_dim2=0
    g1[5] = N_DIM;                                        // tensor_dim0_stride[31:0]
    g1[6] = 0;                                            // stride0 hi | stride1 lo
    g1[7] = 0;
    v4i gz4 = {0, 0, 0, 0};
    v8i gz8 = {0, 0, 0, 0, 0, 0, 0, 0};
    __builtin_amdgcn_tensor_load_to_lds(g0, g1, gz4, gz4, gz8, 0);
}

// ---------------- Kernel 0: xa = x @ lora_A^T  (128 x 16, f32) ----------------
__global__ void __launch_bounds__(256)
qlora_prep_xa(const float* __restrict__ x, const float* __restrict__ loraA,
              float* __restrict__ xa) {
    __shared__ float red[256];
    const int m = blockIdx.x;
    const int tid = threadIdx.x;
    float acc[RLORA];
#pragma unroll
    for (int r = 0; r < RLORA; ++r) acc[r] = 0.0f;
    for (int i = tid; i < K_DIM; i += 256) {
        const float xv = x[m * K_DIM + i];
#pragma unroll
        for (int r = 0; r < RLORA; ++r) acc[r] += xv * loraA[r * K_DIM + i];
    }
    for (int r = 0; r < RLORA; ++r) {
        red[tid] = acc[r];
        __syncthreads();
        for (int off = 128; off > 0; off >>= 1) {
            if (tid < off) red[tid] += red[tid + off];
            __syncthreads();
        }
        if (tid == 0) xa[m * RLORA + r] = red[0];
        __syncthreads();
    }
}

// ---------------- Kernel 1: x (f32) -> xh (f16) ----------------
__global__ void __launch_bounds__(256)
qlora_prep_xh(const float* __restrict__ x, _Float16* __restrict__ xh) {
    const size_t base = ((size_t)blockIdx.x * 256 + threadIdx.x) * 32;
#pragma unroll
    for (int i = 0; i < 32; i += 8) {
        const float4 a = *(const float4*)(x + base + i);
        const float4 b = *(const float4*)(x + base + i + 4);
        union { _Float16 h[8]; uint4 u; } p;
        p.h[0] = (_Float16)a.x; p.h[1] = (_Float16)a.y;
        p.h[2] = (_Float16)a.z; p.h[3] = (_Float16)a.w;
        p.h[4] = (_Float16)b.x; p.h[5] = (_Float16)b.y;
        p.h[6] = (_Float16)b.z; p.h[7] = (_Float16)b.w;
        *(uint4*)(xh + base + i) = p.u;
    }
}

// ---------------- Kernel 2: main fused QLoRA GEMM (TDM double-buffered) ----------------
__global__ void __launch_bounds__(256)
qlora_main(const _Float16* __restrict__ xh, const int* __restrict__ qw,
           const int* __restrict__ zeros, const float* __restrict__ scales,
           const float* __restrict__ loraB, const float* __restrict__ xa,
           float* __restrict__ out) {
    __shared__ _Float16     XL[M_DIM * XSTR];      // 18 KB: x chunk f16 [m][k]
    __shared__ _Float16     WL[NB * WSTR];         //  9 KB: dequant W f16 [n][k]
    __shared__ unsigned int RAW[2][RAWELEM];       // 32 KB: raw int32 codes [k][n], dbl-buf

    const int tid   = threadIdx.x;
    const int wave  = tid >> 5;
    const int lane  = tid & 31;
    const int half  = lane >> 4;
    const int l16   = lane & 15;
    const int n0    = blockIdx.x * NB;
    const int mbase = wave * 16;

    // ---- prologue: wave0 kicks TDM for chunks 0 and 1 ----
    if (wave == 0) {
        tdm_load_w(qw, 0,  n0, (unsigned)(size_t)&RAW[0][0]);
        tdm_load_w(qw, KC, n0, (unsigned)(size_t)&RAW[1][0]);
    }

    // ---- LoRA operands staged into XL/WL (k = 0..31), 4 WMMAs ----
    {
        const int m  = tid >> 1;
        const int kh = (tid & 1) * 16;
#pragma unroll
        for (int k = 0; k < 16; ++k) {
            float v = (kh == 0) ? xa[m * RLORA + k] : 0.0f;
            XL[m * XSTR + kh + k] = (_Float16)v;
        }
        const int nl = tid >> 2;
        const int kq = (tid & 3) * 8;
#pragma unroll
        for (int k = 0; k < 8; ++k) {
            const int kk = kq + k;
            float v = (kk < RLORA) ? SCALING * loraB[(n0 + nl) * RLORA + kk] : 0.0f;
            WL[nl * WSTR + kk] = (_Float16)v;
        }
    }
    __syncthreads();

    v8f c0 = {}, c1 = {}, c2 = {}, c3 = {};
    {
        const v16h a = frag_a(&XL[(mbase + l16) * XSTR], 0, half);
        v16h b = frag_b(&WL[(0  + l16) * WSTR], 0, half);
        c0 = __builtin_amdgcn_wmma_f32_16x16x32_f16(false, a, false, b, (short)0, c0, false, false);
        b = frag_b(&WL[(16 + l16) * WSTR], 0, half);
        c1 = __builtin_amdgcn_wmma_f32_16x16x32_f16(false, a, false, b, (short)0, c1, false, false);
        b = frag_b(&WL[(32 + l16) * WSTR], 0, half);
        c2 = __builtin_amdgcn_wmma_f32_16x16x32_f16(false, a, false, b, (short)0, c2, false, false);
        b = frag_b(&WL[(48 + l16) * WSTR], 0, half);
        c3 = __builtin_amdgcn_wmma_f32_16x16x32_f16(false, a, false, b, (short)0, c3, false, false);
    }

    // ---- main pipelined K loop ----
    for (int c = 0; c < NCHUNK; ++c) {
        const int k0  = c * KC;
        const int buf = c & 1;

        if (wave == 0) {                       // chunk c landed (in-order TDM completion)
            if (c < NCHUNK - 1) __builtin_amdgcn_s_wait_tensorcnt(1);
            else                __builtin_amdgcn_s_wait_tensorcnt(0);
        }
        __syncthreads();                       // RAW[buf] valid; prev compute done (WL/XL free)

        // dequantize RAW[k][n] -> WL[n][k] (f16): 4x4 tile per thread
        {
            const int n4 = (tid & 15) * 4;
            const int k4 = (tid >> 4) * 4;
            const int g  = k0 >> 7;
            const float4 sv = *(const float4*)&scales[(size_t)g * N_DIM + n0 + n4];
            const int4   zv = *(const int4*)&zeros[(size_t)g * N_DIM + n0 + n4];
            const float zs0 = sv.x * (float)zv.x, zs1 = sv.y * (float)zv.y;
            const float zs2 = sv.z * (float)zv.z, zs3 = sv.w * (float)zv.w;
            _Float16 w[4][4];                  // [n][k]
#pragma unroll
            for (int i = 0; i < 4; ++i) {      // k
                const uint4 q = *(const uint4*)&RAW[buf][(k4 + i) * NB + n4];
                w[0][i] = (_Float16)fmaf((float)(int)q.x, sv.x, -zs0);
                w[1][i] = (_Float16)fmaf((float)(int)q.y, sv.y, -zs1);
                w[2][i] = (_Float16)fmaf((float)(int)q.z, sv.z, -zs2);
                w[3][i] = (_Float16)fmaf((float)(int)q.w, sv.w, -zs3);
            }
#pragma unroll
            for (int j = 0; j < 4; ++j) {
                union { _Float16 h[4]; unsigned long long u; } p;
                p.h[0] = w[j][0]; p.h[1] = w[j][1]; p.h[2] = w[j][2]; p.h[3] = w[j][3];
                *(unsigned long long*)&WL[(n4 + j) * WSTR + k4] = p.u;
            }
        }
        // stage x chunk (already f16, L2-resident): pure b128 copies
        {
            const int m  = tid >> 1;
            const int kh = (tid & 1) * 32;
            const uint4* src = (const uint4*)(xh + (size_t)m * K_DIM + k0 + kh);
            uint4* dst = (uint4*)&XL[m * XSTR + kh];
            dst[0] = src[0]; dst[1] = src[1]; dst[2] = src[2]; dst[3] = src[3];
        }
        __syncthreads();                       // WL/XL ready; RAW[buf] fully consumed

        if (wave == 0 && c + 2 < NCHUNK)       // refill freed RAW buffer
            tdm_load_w(qw, (c + 2) * KC, n0, (unsigned)(size_t)&RAW[buf][0]);

        // 8 WMMAs: 2 k-steps x 4 n-tiles
        const _Float16* arow = &XL[(mbase + l16) * XSTR];
#pragma unroll
        for (int kk = 0; kk < KC; kk += 32) {
            const v16h a = frag_a(arow, kk, half);
            v16h b = frag_b(&WL[(0  + l16) * WSTR], kk, half);
            c0 = __builtin_amdgcn_wmma_f32_16x16x32_f16(false, a, false, b, (short)0, c0, false, false);
            b = frag_b(&WL[(16 + l16) * WSTR], kk, half);
            c1 = __builtin_amdgcn_wmma_f32_16x16x32_f16(false, a, false, b, (short)0, c1, false, false);
            b = frag_b(&WL[(32 + l16) * WSTR], kk, half);
            c2 = __builtin_amdgcn_wmma_f32_16x16x32_f16(false, a, false, b, (short)0, c2, false, false);
            b = frag_b(&WL[(48 + l16) * WSTR], kk, half);
            c3 = __builtin_amdgcn_wmma_f32_16x16x32_f16(false, a, false, b, (short)0, c3, false, false);
        }
    }

    // ---- epilogue: C layout (lane owns column n; VGPR j -> row j + 8*half) ----
#pragma unroll
    for (int j = 0; j < 8; ++j) {
        const int m = mbase + j + 8 * half;
        float* orow = &out[(size_t)m * N_DIM + n0 + l16];
        orow[0]  = c0[j];
        orow[16] = c1[j];
        orow[32] = c2[j];
        orow[48] = c3[j];
    }
}

extern "C" void kernel_launch(void* const* d_in, const int* in_sizes, int n_in,
                              void* d_out, int out_size, void* d_ws, size_t ws_size,
                              hipStream_t stream) {
    const float* x      = (const float*)d_in[0];
    const int*   qw     = (const int*)d_in[1];
    const int*   zeros  = (const int*)d_in[2];
    const float* scales = (const float*)d_in[3];
    const float* loraA  = (const float*)d_in[4];
    const float* loraB  = (const float*)d_in[5];
    float*       out    = (float*)d_out;

    float*     xa = (float*)d_ws;                          // 8 KB
    _Float16*  xh = (_Float16*)((char*)d_ws + 8192);       // 1 MB f16 copy of x

    qlora_prep_xa<<<M_DIM, 256, 0, stream>>>(x, loraA, xa);
    qlora_prep_xh<<<(M_DIM * K_DIM) / (256 * 32), 256, 0, stream>>>(x, xh);
    qlora_main<<<N_DIM / NB, 256, 0, stream>>>(xh, qw, zeros, scales, loraB, xa, out);
}
